// AblationModel_79422535238095
// MI455X (gfx1250) — matrix operand order repaired and checked by reading, not compiled, not run on hardware
//
#include <hip/hip_runtime.h>
#include <hip/hip_bf16.h>

// ---------------------------------------------------------------------------
// CDNA5 (gfx1250): convs + FC layers via v_wmma_f32_16x16x32_f16 (wave32).
// Block tile 64(M) x 64(N): 4 waves x 4 M-subtiles -> 16 WMMA tiles per block,
// 4 WMMAs per wave per K-step sharing one B fragment (4x B reuse).
// Tile staging via GLOBAL_LOAD_ASYNC_TO_LDS_B128 (ASYNCcnt) on the fast path.
// ---------------------------------------------------------------------------

typedef _Float16 half_t;
typedef __attribute__((ext_vector_type(16))) _Float16 v16h;
typedef __attribute__((ext_vector_type(8)))  float    v8f;

__device__ __forceinline__ int imin(int a, int b){ return a < b ? a : b; }
__device__ __forceinline__ int imax(int a, int b){ return a > b ? a : b; }
__device__ __forceinline__ float sigf(float v){ return 1.f/(1.f+__expf(-v)); }
__device__ __forceinline__ float actf(float v, int act){
  if (act == 1) return v > 0.f ? v : 0.f;
  if (act == 2) return sigf(v);
  return v;
}

// Async global->LDS copy of 16 bytes per lane (GV addressing, ASYNCcnt).
__device__ __forceinline__ void async_b128(const float* gaddr, unsigned lds_off){
  asm volatile("global_load_async_to_lds_b128 %0, %1, off"
               :: "v"(lds_off), "v"(gaddr) : "memory");
}
__device__ __forceinline__ void wait_async(){
  asm volatile("s_wait_asynccnt 0x0" ::: "memory");
}
__device__ __forceinline__ unsigned lds_addr(const void* p){
  return (unsigned)(unsigned long long)p;
}

// Build a v16h fragment from two runs of 8 f32 in LDS (ds_load_b128 + cvt).
__device__ __forceinline__ v16h cvt16(const float* p0, const float* p1){
  v16h r;
  #pragma unroll
  for (int i = 0; i < 8; ++i) { r[i] = (half_t)p0[i]; r[i+8] = (half_t)p1[i]; }
  return r;
}

// ------------------------- generic GEMM: C = act(A * W^T + b1 + b2) --------
// A: [M,K] f32 row-major; W: [N,K] f32 row-major; C: [M,N] f32.
// Block = 128 threads = 4 waves; tile 64(M) x 64(N); wave w owns N-subtile w
// and accumulates 4 M-subtiles.
__global__ __launch_bounds__(128) void gemm_wmma_kernel(
    const float* __restrict__ A, const float* __restrict__ Bw,
    float* __restrict__ C, int M, int N, int K,
    const float* __restrict__ bias1, const float* __restrict__ bias2, int act)
{
  __shared__ alignas(16) float lAf[64][36];      // [m][k] f32, 144B row stride
  __shared__ alignas(16) float lBf[4][16][36];   // [wave][n][k] f32
  const int tid  = threadIdx.x;
  const int wv   = tid >> 5;
  const int lane = tid & 31;
  const int tm   = blockIdx.x * 64;
  const int tn   = blockIdx.y * 64 + wv * 16;
  const bool kAligned = (K & 3) == 0;

  v8f acc[4] = {};
  for (int k0 = 0; k0 < K; k0 += 32) {
    __syncthreads();
    const bool fastA = kAligned && (k0 + 32 <= K) && (tm + 64 <= M);
    const bool fastB = kAligned && (k0 + 32 <= K) && (tn + 16 <= N);

    if (fastA) {
      // 64 rows x 128B = 512 chunks of 16B; 4 per thread.
      #pragma unroll
      for (int j = 0; j < 4; ++j) {
        int chk = tid + j * 128;
        int row = chk >> 3, off = (chk & 7) * 4;
        async_b128(A + (size_t)(tm + row) * K + k0 + off,
                   lds_addr(&lAf[row][off]));
      }
    } else {
      // rows wv+4i, column = lane (clamped unconditional loads)
      #pragma unroll
      for (int i = 0; i < 16; ++i) {
        int mi  = wv + 4 * i;
        int row = tm + mi, col = k0 + lane;
        int rc = row < M ? row : M - 1;
        int cc = col < K ? col : K - 1;
        float v = A[(size_t)rc * K + cc];
        lAf[mi][lane] = (row < M && col < K) ? v : 0.f;
      }
    }
    if (fastB) {
      #pragma unroll
      for (int j = 0; j < 4; ++j) {
        int chk = lane + j * 32;
        int row = chk >> 3, off = (chk & 7) * 4;
        async_b128(Bw + (size_t)(tn + row) * K + k0 + off,
                   lds_addr(&lBf[wv][row][off]));
      }
    } else {
      #pragma unroll
      for (int i = 0; i < 16; ++i) {
        int e  = lane + i * 32;
        int kk = e >> 4, nn = e & 15;
        int n = tn + nn, col = k0 + kk;
        int nc = n < N ? n : N - 1;
        int cc = col < K ? col : K - 1;
        float v = Bw[(size_t)nc * K + cc];
        lBf[wv][nn][kk] = (n < N && col < K) ? v : 0.f;
      }
    }
    if (k0 + 32 < K) {
      int n = tn + (lane & 15);
      if (n < N) __builtin_prefetch(Bw + (size_t)n * K + k0 + 32, 0, 1);
    }
    wait_async();
    __syncthreads();

    // B frag: n = lane&15; lanes 0-15 -> K {0..7,16..23}; lanes 16-31 -> K {8..15,24..31}
    const int bn = lane & 15;
    const int bk = (lane < 16) ? 0 : 8;
    v16h b = cvt16(&lBf[wv][bn][bk], &lBf[wv][bn][bk + 16]);
    // A frags: lanes 0-15 -> K {0..7,16..23}; lanes 16-31 -> K {8..15,24..31}
    const int am = lane & 15;
    const int ak = (lane < 16) ? 0 : 8;
    #pragma unroll
    for (int ms = 0; ms < 4; ++ms) {
      v16h a = cvt16(&lAf[ms * 16 + am][ak], &lAf[ms * 16 + am][ak + 16]);
      acc[ms] = __builtin_amdgcn_wmma_f32_16x16x32_f16(false, a, false, b,
                                                       (short)0, acc[ms],
                                                       false, false);
    }
  }

  // Store: VGPR r -> M = r + (lane<16 ? 0 : 8); N = lane & 15
  const int n = tn + (lane & 15);
  if (n < N) {
    float bv = (bias1 ? bias1[n] : 0.f) + (bias2 ? bias2[n] : 0.f);
    #pragma unroll
    for (int ms = 0; ms < 4; ++ms) {
      int mb = tm + ms * 16 + ((lane >> 4) << 3);
      #pragma unroll
      for (int r = 0; r < 8; ++r) {
        int m = mb + r;
        if (m < M) C[(size_t)m * N + n] = actf(acc[ms][r] + bv, act);
      }
    }
  }
}

// ------------- implicit-im2col conv + bias + BN + ReLU, via WMMA -----------
// in: [B,Cin,IH,IW]; wgt: [Cout,Cin,KH,KW]; out: [B,Cout,OH,OW]
// Requires M = B*OH*OW divisible by 64 (true for all calls here).
__global__ __launch_bounds__(128) void conv_wmma_kernel(
    const float* __restrict__ in, const float* __restrict__ wgt,
    float* __restrict__ out,
    int Bn, int Cin, int IH, int IW, int Cout,
    int KH, int KW, int stride, int pad, int OH, int OW,
    const float* __restrict__ bias, const float* __restrict__ gamma,
    const float* __restrict__ beta)
{
  __shared__ alignas(16) float lAf[64][36];
  __shared__ alignas(16) float lBf[4][16][36];
  const int tid  = threadIdx.x;
  const int wv   = tid >> 5;
  const int lane = tid & 31;
  const int tm   = blockIdx.x * 64;
  const int tn   = blockIdx.y * 64 + wv * 16;
  const int khw  = KH * KW;
  const int Kc   = Cin * khw;
  const int ohw  = OH * OW;
  const int chw  = Cin * IH * IW;
  const bool kAligned = (Kc & 3) == 0;

  // Hoist per-row im2col decomposition out of the K loop:
  // this thread stages rows m = tm + wv + 4*i at column kk = lane.
  int rbase[16], riy0[16], rix0[16];
  #pragma unroll
  for (int i = 0; i < 16; ++i) {
    int m  = tm + wv + 4 * i;
    int bi = m / ohw; int r2 = m - bi * ohw;
    int oy = r2 / OW; int ox = r2 - oy * OW;
    rbase[i] = bi * chw;
    riy0[i]  = oy * stride - pad;
    rix0[i]  = ox * stride - pad;
  }

  v8f acc[4] = {};
  for (int k0 = 0; k0 < Kc; k0 += 32) {
    __syncthreads();
    // A: im2col gather. One k-decomposition per thread per step (kk == lane).
    {
      int col = k0 + lane;
      bool colok = col < Kc;
      int cc  = colok ? col : Kc - 1;
      int cin = cc / khw; int r3 = cc - cin * khw;
      int ky  = r3 / KW;  int kx = r3 - ky * KW;
      int cinoff = cin * IH * IW;
      #pragma unroll
      for (int i = 0; i < 16; ++i) {
        int iy = riy0[i] + ky, ix = rix0[i] + kx;
        bool ok = colok && iy >= 0 && iy < IH && ix >= 0 && ix < IW;
        int iyc = imin(imax(iy, 0), IH - 1);
        int ixc = imin(imax(ix, 0), IW - 1);
        float v = in[(size_t)(rbase[i] + cinoff + iyc * IW + ixc)];
        lAf[wv + 4 * i][lane] = ok ? v : 0.f;
      }
    }
    // B: weights dense [Cout, Kc] -> async fast path when aligned
    const bool fastB = kAligned && (k0 + 32 <= Kc) && (tn + 16 <= Cout);
    if (fastB) {
      #pragma unroll
      for (int j = 0; j < 4; ++j) {
        int chk = lane + j * 32;
        int row = chk >> 3, off = (chk & 7) * 4;
        async_b128(wgt + (size_t)(tn + row) * Kc + k0 + off,
                   lds_addr(&lBf[wv][row][off]));
      }
    } else {
      #pragma unroll
      for (int i = 0; i < 16; ++i) {
        int e  = lane + i * 32;
        int kk = e >> 4, nn = e & 15;
        int n = tn + nn, col = k0 + kk;
        int nc = n < Cout ? n : Cout - 1;
        int cc = col < Kc ? col : Kc - 1;
        float v = wgt[(size_t)nc * Kc + cc];
        lBf[wv][nn][kk] = (n < Cout && col < Kc) ? v : 0.f;
      }
    }
    wait_async();
    __syncthreads();

    const int bn = lane & 15;
    const int bk = (lane < 16) ? 0 : 8;
    v16h b = cvt16(&lBf[wv][bn][bk], &lBf[wv][bn][bk + 16]);
    const int am = lane & 15;
    const int ak = (lane < 16) ? 0 : 8;
    #pragma unroll
    for (int ms = 0; ms < 4; ++ms) {
      v16h a = cvt16(&lAf[ms * 16 + am][ak], &lAf[ms * 16 + am][ak + 16]);
      acc[ms] = __builtin_amdgcn_wmma_f32_16x16x32_f16(false, a, false, b,
                                                       (short)0, acc[ms],
                                                       false, false);
    }
  }

  const int n = tn + (lane & 15);
  if (n < Cout) {
    float bv = bias[n], gv = gamma[n], btv = beta[n];
    #pragma unroll
    for (int ms = 0; ms < 4; ++ms) {
      int mb = tm + ms * 16 + ((lane >> 4) << 3);
      #pragma unroll
      for (int r = 0; r < 8; ++r) {
        int m = mb + r;
        int bi = m / ohw; int r2 = m - bi * ohw;
        float v = (acc[ms][r] + bv) * gv + btv;
        out[((size_t)bi * Cout + n) * ohw + r2] = v > 0.f ? v : 0.f;
      }
    }
  }
}

// ------------------------------- glue kernels ------------------------------
__global__ void zero_kernel(float* p, int n){
  int i = blockIdx.x * blockDim.x + threadIdx.x;
  if (i < n) p[i] = 0.f;
}

__global__ void maxpool2_kernel(const float* __restrict__ in, float* __restrict__ out,
                                int N, int C, int H, int W){
  int idx = blockIdx.x * blockDim.x + threadIdx.x;
  int OH = H >> 1, OW = W >> 1;
  int total = N * C * OH * OW;
  if (idx >= total) return;
  int ox = idx % OW; int t = idx / OW;
  int oy = t % OH;   t /= OH;
  int c  = t % C;    int n = t / C;
  const float* p = in + (((size_t)n * C + c) * H + oy * 2) * W + ox * 2;
  out[idx] = fmaxf(fmaxf(p[0], p[1]), fmaxf(p[W], p[W + 1]));
}

// y[b][c][p]: p<H -> mean over W at row p; p>=H -> mean over H at col p-H
__global__ void pool_hw_kernel(const float* __restrict__ x, float* __restrict__ y,
                               int C, int H, int W){
  int bc = blockIdx.x;
  int p  = threadIdx.x;
  const float* xp = x + (size_t)bc * H * W;
  float s = 0.f;
  if (p < H) { for (int w = 0; w < W; ++w) s += xp[p * W + w]; s /= (float)W; }
  else       { int w = p - H; for (int h = 0; h < H; ++h) s += xp[h * W + w]; s /= (float)H; }
  y[(size_t)bc * (H + W) + p] = s;
}

// y2[b][m][p] = relu(bn(sum_c w[m,c]*y[b,c,p] + bias[m]))   (mip = 8)
__global__ void att_mid_kernel(const float* __restrict__ y, const float* __restrict__ w,
                               const float* __restrict__ bi, const float* __restrict__ g,
                               const float* __restrict__ bt, float* __restrict__ y2,
                               int Bn, int C, int P){
  int idx = blockIdx.x * blockDim.x + threadIdx.x;
  int total = Bn * 8 * P;
  if (idx >= total) return;
  int p = idx % P; int t = idx / P;
  int m = t % 8;   int b = t / 8;
  const float* yb = y + (size_t)b * C * P + p;
  const float* wm = w + (size_t)m * C;
  float s = bi[m];
  for (int c = 0; c < C; ++c) s += wm[c] * yb[(size_t)c * P];
  s = s * g[m] + bt[m];
  y2[idx] = s > 0.f ? s : 0.f;
}

// gate[b][co][p] = sigmoid(sum_m w[co,m]*y2[b,m,off+p] + bias[co])
__global__ void att_gate_kernel(const float* __restrict__ y2, const float* __restrict__ w,
                                const float* __restrict__ bi, float* __restrict__ gate,
                                int Bn, int C, int P, int off, int Ptot){
  int idx = blockIdx.x * blockDim.x + threadIdx.x;
  int total = Bn * C * P;
  if (idx >= total) return;
  int p  = idx % P; int t = idx / P;
  int co = t % C;   int b = t / C;
  float s = bi[co];
  #pragma unroll
  for (int m = 0; m < 8; ++m)
    s += w[co * 8 + m] * y2[((size_t)b * 8 + m) * Ptot + off + p];
  gate[idx] = sigf(s);
}

__global__ void att_apply_kernel(float* __restrict__ x, const float* __restrict__ ah,
                                 const float* __restrict__ aw, int Bn, int C, int H, int W){
  int idx = blockIdx.x * blockDim.x + threadIdx.x;
  int total = Bn * C * H * W;
  if (idx >= total) return;
  int w = idx % W; int t = idx / W;
  int h = t % H;   t /= H;
  int c = t % C;   int b = t / C;
  size_t bc = (size_t)b * C + c;
  x[idx] *= ah[bc * H + h] * aw[bc * W + w];
}

__global__ void gmean_pool_kernel(const float* __restrict__ x, float* __restrict__ out,
                                  int HW, int total){
  int idx = blockIdx.x * blockDim.x + threadIdx.x;
  if (idx >= total) return;
  const float* xp = x + (size_t)idx * HW;
  float s = 0.f;
  for (int i = 0; i < HW; ++i) s += xp[i];
  out[idx] = s / (float)HW;
}

// 32x32 -> 4x4 by 8x8 mean; out[n][c*16 + i*4 + j]
__global__ void focus_pool_kernel(const float* __restrict__ x, float* __restrict__ out,
                                  int C, int total){
  int idx = blockIdx.x * blockDim.x + threadIdx.x;
  if (idx >= total) return;
  int j = idx & 3; int t = idx >> 2;
  int i = t & 3;   t >>= 2;
  int c = t % C;   int n = t / C;
  const float* xp = x + ((size_t)n * C + c) * 1024;
  float s = 0.f;
  for (int a = 0; a < 8; ++a)
    for (int b2 = 0; b2 < 8; ++b2)
      s += xp[(i * 8 + a) * 32 + j * 8 + b2];
  out[(size_t)n * (C * 16) + c * 16 + i * 4 + j] = s * (1.f / 64.f);
}

__device__ __forceinline__ void axis_coords(float cpos, int full, int j,
                                            int& i0g, int& i1g, float& frac){
  int ci = (int)floorf(cpos * (float)full);
  int lo = imax(0, ci - 32);           // PS/2 = 32
  int hi = imin(full, lo + 64);        // PS = 64
  int w  = hi - lo;
  float s = fmaxf((j + 0.5f) * (float)w / 64.f - 0.5f, 0.f);
  int i0 = (int)floorf(s);
  frac = s - (float)i0;
  i0 = imin(i0, w - 1);
  int i1 = imin(i0 + 1, w - 1);
  i0g = lo + i0; i1g = lo + i1;
}

// patches[b][t][c][jy][jx], B=32 T=8 HW=256 OUT=64
__global__ void crop_resize_kernel(const float* __restrict__ images,
                                   const float* __restrict__ paths,
                                   float* __restrict__ patches){
  int idx = blockIdx.x * blockDim.x + threadIdx.x;
  const int total = 32 * 8 * 3 * 64 * 64;
  if (idx >= total) return;
  int jx = idx & 63; int t = idx >> 6;
  int jy = t & 63;   t >>= 6;
  int c  = t % 3;    t /= 3;
  int tt = t % 8;    int b = t / 8;
  float px = paths[((size_t)b * 8 + tt) * 2 + 0];
  float py = paths[((size_t)b * 8 + tt) * 2 + 1];
  int gx0, gx1, gy0, gy1; float fx, fy;
  axis_coords(px, 256, jx, gx0, gx1, fx);
  axis_coords(py, 256, jy, gy0, gy1, fy);
  const float* img = images + ((size_t)b * 3 + c) * 65536;
  float top = (1.f - fx) * img[gy0 * 256 + gx0] + fx * img[gy0 * 256 + gx1];
  float bot = (1.f - fx) * img[gy1 * 256 + gx0] + fx * img[gy1 * 256 + gx1];
  patches[idx] = (1.f - fy) * top + fy * bot;
}

// pe[r][m] = relu(gt[r,0]*w[m,0] + gt[r,1]*w[m,1] + b[m])
__global__ void pe1_kernel(const float* __restrict__ gt, const float* __restrict__ w,
                           const float* __restrict__ b, float* __restrict__ pe){
  int idx = blockIdx.x * blockDim.x + threadIdx.x;
  if (idx >= 256 * 256) return;
  int m = idx & 255; int r = idx >> 8;
  float s = gt[r * 2] * w[m * 2] + gt[r * 2 + 1] * w[m * 2 + 1] + b[m];
  pe[idx] = s > 0.f ? s : 0.f;
}

__global__ void feat_add_kernel(const float* __restrict__ gfeat, const float* __restrict__ loc,
                                const float* __restrict__ posf, float* __restrict__ feat){
  int idx = blockIdx.x * blockDim.x + threadIdx.x;
  if (idx >= 256 * 512) return;
  int j = idx & 511; int r = idx >> 9; int b = r >> 3;
  feat[idx] = gfeat[(b << 9) + j] + loc[idx] + posf[idx];
}

__global__ void lstm_gate_kernel(const float* __restrict__ Zx, const float* __restrict__ Zh,
                                 float* __restrict__ h, float* __restrict__ c,
                                 float* __restrict__ outh, int t){
  int idx = blockIdx.x * blockDim.x + threadIdx.x;
  if (idx >= 32 * 512) return;
  int j = idx & 511; int b = idx >> 9;
  const float* zx = Zx + ((size_t)b * 8 + t) * 2048;
  const float* zh = Zh + (size_t)b * 2048;
  float zi = zx[j]        + zh[j];
  float zf = zx[512 + j]  + zh[512 + j];
  float zg = zx[1024 + j] + zh[1024 + j];
  float zo = zx[1536 + j] + zh[1536 + j];
  float cn = sigf(zf) * c[idx] + sigf(zi) * tanhf(zg);
  float hn = sigf(zo) * tanhf(cn);
  c[idx] = cn; h[idx] = hn;
  outh[((size_t)b * 8 + t) * 512 + j] = hn;
}

// ---------------------------------------------------------------------------
extern "C" void kernel_launch(void* const* d_in, const int* in_sizes, int n_in,
                              void* d_out, int out_size, void* d_ws, size_t ws_size,
                              hipStream_t stream) {
  // Inputs: images, saliency_map(unused), gt_scanpaths, then params in
  // jax-pytree (sorted dict key) order.
  const float* images = (const float*)d_in[0];
  const float* gt     = (const float*)d_in[2];
  const float* d1_b   = (const float*)d_in[3];
  const float* d1_w   = (const float*)d_in[4];
  const float* d2_b   = (const float*)d_in[5];
  const float* d2_w   = (const float*)d_in[6];
  const float* f_bn1_b = (const float*)d_in[7];
  const float* f_bn1_g = (const float*)d_in[8];
  const float* f_bn2_b = (const float*)d_in[9];
  const float* f_bn2_g = (const float*)d_in[10];
  const float* f_c1_b  = (const float*)d_in[11];
  const float* f_c1_w  = (const float*)d_in[12];
  const float* f_c2_b  = (const float*)d_in[13];
  const float* f_c2_w  = (const float*)d_in[14];
  const float* f_ca_bn_b = (const float*)d_in[15];
  const float* f_ca_bn_g = (const float*)d_in[16];
  const float* f_ca_c1_b = (const float*)d_in[17];
  const float* f_ca_c1_w = (const float*)d_in[18];
  const float* f_ca_ch_b = (const float*)d_in[19];
  const float* f_ca_ch_w = (const float*)d_in[20];
  const float* f_ca_cw_b = (const float*)d_in[21];
  const float* f_ca_cw_w = (const float*)d_in[22];
  const float* f_fc_b  = (const float*)d_in[23];
  const float* f_fc_w  = (const float*)d_in[24];
  const float* g_bn1_b = (const float*)d_in[25];
  const float* g_bn1_g = (const float*)d_in[26];
  const float* g_bn2_b = (const float*)d_in[27];
  const float* g_bn2_g = (const float*)d_in[28];
  const float* g_bn3_b = (const float*)d_in[29];
  const float* g_bn3_g = (const float*)d_in[30];
  const float* g_c1_b  = (const float*)d_in[31];
  const float* g_c1_w  = (const float*)d_in[32];
  const float* g_c2_b  = (const float*)d_in[33];
  const float* g_c2_w  = (const float*)d_in[34];
  const float* g_c3_b  = (const float*)d_in[35];
  const float* g_c3_w  = (const float*)d_in[36];
  const float* g_ca1_bn_b = (const float*)d_in[37];
  const float* g_ca1_bn_g = (const float*)d_in[38];
  const float* g_ca1_c1_b = (const float*)d_in[39];
  const float* g_ca1_c1_w = (const float*)d_in[40];
  const float* g_ca1_ch_b = (const float*)d_in[41];
  const float* g_ca1_ch_w = (const float*)d_in[42];
  const float* g_ca1_cw_b = (const float*)d_in[43];
  const float* g_ca1_cw_w = (const float*)d_in[44];
  const float* g_ca2_bn_b = (const float*)d_in[45];
  const float* g_ca2_bn_g = (const float*)d_in[46];
  const float* g_ca2_c1_b = (const float*)d_in[47];
  const float* g_ca2_c1_w = (const float*)d_in[48];
  const float* g_ca2_ch_b = (const float*)d_in[49];
  const float* g_ca2_ch_w = (const float*)d_in[50];
  const float* g_ca2_cw_b = (const float*)d_in[51];
  const float* g_ca2_cw_w = (const float*)d_in[52];
  const float* g_fc_b  = (const float*)d_in[53];
  const float* g_fc_w  = (const float*)d_in[54];
  const float* lstm_bhh = (const float*)d_in[55];
  const float* lstm_bih = (const float*)d_in[56];
  const float* lstm_whh = (const float*)d_in[57];
  const float* lstm_wih = (const float*)d_in[58];
  const float* pe1_b   = (const float*)d_in[59];
  const float* pe1_w   = (const float*)d_in[60];
  const float* pe2_b   = (const float*)d_in[61];
  const float* pe2_w   = (const float*)d_in[62];

  // ---- workspace arenas (floats) ----
  float* ws = (float*)d_ws;
  size_t off = 0;
  auto take = [&](size_t n){ float* p = ws + off; off += n; return p; };
  float* big0    = take(16777216);  // conv1 chunk scratch
  float* gpbuf   = take(8388608);   // glance pool1 / focus pool chunk
  float* g2buf   = take(8388608);   // glance conv2 / focus conv2 chunk
  float* g3buf   = take(2097152);   // glance conv3 [32,256,16,16]
  float* atty    = take(524288);
  float* atty2   = take(32768);
  float* attah   = take(262144);
  float* attaw   = take(262144);
  float* gmean   = take(8192);      // [32,256]
  float* gfeat   = take(16384);     // [32,512]
  float* patches = take(3145728);   // [256,3,64,64]
  float* pooledf = take(524288);    // [256,2048]
  float* localb  = take(131072);    // [256,512]
  float* peb     = take(65536);     // [256,256]
  float* posf    = take(131072);    // [256,512]
  float* featb   = take(131072);    // [256,512]
  float* Zx      = take(524288);    // [256,2048]
  float* Zh      = take(65536);     // [32,2048]
  float* hstate  = take(16384);     // [32,512]
  float* cstate  = take(16384);     // [32,512] (contiguous after hstate)
  float* lstm_o  = take(131072);    // [32,8,512]
  float* hid     = take(65536);     // [256,256]

  auto gemm = [&](const float* A, const float* W, float* C, int M, int N, int K,
                  const float* b1, const float* b2, int act){
    dim3 g((M + 63) / 64, (N + 63) / 64);
    gemm_wmma_kernel<<<g, dim3(128), 0, stream>>>(A, W, C, M, N, K, b1, b2, act);
  };
  auto conv = [&](const float* in, const float* w, float* out,
                  int Bn, int Cin, int IH, int IW, int Cout, int KH, int KW,
                  int st, int pad, int OH, int OW,
                  const float* b, const float* gma, const float* bta){
    int M = Bn * OH * OW;             // divisible by 64 for every call below
    dim3 g(M / 64, (Cout + 63) / 64);
    conv_wmma_kernel<<<g, dim3(128), 0, stream>>>(in, w, out, Bn, Cin, IH, IW,
                                                  Cout, KH, KW, st, pad, OH, OW,
                                                  b, gma, bta);
  };
  auto coord_att = [&](float* x, int Bn, int C, int H, int W,
                       const float* c1w, const float* c1b, const float* bng,
                       const float* bnb, const float* chw, const float* chb,
                       const float* cww, const float* cwb){
    int P = H + W;
    pool_hw_kernel<<<Bn * C, P, 0, stream>>>(x, atty, C, H, W);
    int t1 = Bn * 8 * P;
    att_mid_kernel<<<(t1 + 255) / 256, 256, 0, stream>>>(atty, c1w, c1b, bng, bnb,
                                                         atty2, Bn, C, P);
    int t2 = Bn * C * H;
    att_gate_kernel<<<(t2 + 255) / 256, 256, 0, stream>>>(atty2, chw, chb, attah,
                                                          Bn, C, H, 0, P);
    int t3 = Bn * C * W;
    att_gate_kernel<<<(t3 + 255) / 256, 256, 0, stream>>>(atty2, cww, cwb, attaw,
                                                          Bn, C, W, H, P);
    int t4 = Bn * C * H * W;
    att_apply_kernel<<<(t4 + 255) / 256, 256, 0, stream>>>(x, attah, attaw, Bn, C, H, W);
  };

  // ---------------- glance ----------------
  const size_t IMG = 3ull * 256 * 256;
  for (int ch = 0; ch < 4; ++ch) {   // 8 images per chunk
    conv(images + (size_t)ch * 8 * IMG, g_c1_w, big0,
         8, 3, 256, 256, 64, 7, 7, 2, 3, 128, 128, g_c1_b, g_bn1_g, g_bn1_b);
    int tot = 8 * 64 * 64 * 64;
    maxpool2_kernel<<<(tot + 255) / 256, 256, 0, stream>>>(big0,
        gpbuf + (size_t)ch * tot, 8, 64, 128, 128);
  }
  conv(gpbuf, g_c2_w, g2buf, 32, 64, 64, 64, 128, 3, 3, 2, 1, 32, 32,
       g_c2_b, g_bn2_g, g_bn2_b);
  coord_att(g2buf, 32, 128, 32, 32, g_ca1_c1_w, g_ca1_c1_b, g_ca1_bn_g, g_ca1_bn_b,
            g_ca1_ch_w, g_ca1_ch_b, g_ca1_cw_w, g_ca1_cw_b);
  conv(g2buf, g_c3_w, g3buf, 32, 128, 32, 32, 256, 3, 3, 2, 1, 16, 16,
       g_c3_b, g_bn3_g, g_bn3_b);
  coord_att(g3buf, 32, 256, 16, 16, g_ca2_c1_w, g_ca2_c1_b, g_ca2_bn_g, g_ca2_bn_b,
            g_ca2_ch_w, g_ca2_ch_b, g_ca2_cw_w, g_ca2_cw_b);
  gmean_pool_kernel<<<(32 * 256 + 255) / 256, 256, 0, stream>>>(g3buf, gmean, 256, 32 * 256);
  gemm(gmean, g_fc_w, gfeat, 32, 512, 256, g_fc_b, nullptr, 0);

  // ---------------- crop + focus ----------------
  {
    int tcr = 32 * 8 * 3 * 64 * 64;
    crop_resize_kernel<<<(tcr + 255) / 256, 256, 0, stream>>>(images, gt, patches);
  }
  for (int ch = 0; ch < 4; ++ch) {   // 64 patches per chunk
    const float* pin = patches + (size_t)ch * 64 * 3 * 64 * 64;
    conv(pin, f_c1_w, big0, 64, 3, 64, 64, 64, 5, 5, 1, 2, 64, 64,
         f_c1_b, f_bn1_g, f_bn1_b);
    int tot = 64 * 64 * 32 * 32;
    maxpool2_kernel<<<(tot + 255) / 256, 256, 0, stream>>>(big0, gpbuf, 64, 64, 64, 64);
    conv(gpbuf, f_c2_w, g2buf, 64, 64, 32, 32, 128, 3, 3, 1, 1, 32, 32,
         f_c2_b, f_bn2_g, f_bn2_b);
    coord_att(g2buf, 64, 128, 32, 32, f_ca_c1_w, f_ca_c1_b, f_ca_bn_g, f_ca_bn_b,
              f_ca_ch_w, f_ca_ch_b, f_ca_cw_w, f_ca_cw_b);
    int tp = 64 * 128 * 16;
    focus_pool_kernel<<<(tp + 255) / 256, 256, 0, stream>>>(g2buf,
        pooledf + (size_t)ch * 64 * 2048, 128, tp);
  }
  gemm(pooledf, f_fc_w, localb, 256, 512, 2048, f_fc_b, nullptr, 0);

  // ---------------- positional + fuse ----------------
  pe1_kernel<<<(256 * 256 + 255) / 256, 256, 0, stream>>>(gt, pe1_w, pe1_b, peb);
  gemm(peb, pe2_w, posf, 256, 512, 256, pe2_b, nullptr, 0);
  feat_add_kernel<<<(256 * 512 + 255) / 256, 256, 0, stream>>>(gfeat, localb, posf, featb);

  // ---------------- LSTM ----------------
  gemm(featb, lstm_wih, Zx, 256, 2048, 512, lstm_bih, lstm_bhh, 0);
  zero_kernel<<<(32768 + 255) / 256, 256, 0, stream>>>(hstate, 32768); // h + c
  for (int t = 0; t < 8; ++t) {
    gemm(hstate, lstm_whh, Zh, 32, 2048, 512, nullptr, nullptr, 0);
    lstm_gate_kernel<<<(32 * 512 + 255) / 256, 256, 0, stream>>>(Zx, Zh, hstate,
                                                                 cstate, lstm_o, t);
  }

  // ---------------- decoder ----------------
  gemm(lstm_o, d1_w, hid, 256, 256, 512, d1_b, nullptr, 1);
  gemm(hid, d2_w, (float*)d_out, 256, 2, 256, d2_b, nullptr, 2);
}
